// conv_encoder_59785944760467
// MI455X (gfx1250) — matrix operand order repaired and checked by reading, not compiled, not run on hardware
//
#include <hip/hip_runtime.h>

typedef __attribute__((ext_vector_type(16))) __bf16 v16bf;
typedef __attribute__((ext_vector_type(8)))  __bf16 v8bf;
typedef __attribute__((ext_vector_type(8)))  float  v8f;

#define NEURONS   4096
#define CHANNELS  64
#define HW        4096
#define BATCH     32
#define KTILE     64
#define NTILE     128
#define LDS_STRIDE 72   // 64 + 8 pad -> 144B rows = 9 x 16B superbanks, gcd(9,16)=1 (conflict-free b128)

#define XN  ((size_t)BATCH * CHANNELS * HW)   // 8,388,608  x elements
#define WSN ((size_t)NEURONS * HW)            // 16,777,216 W_s elements
#define WS_BYTES ((2 * XN + 2 * WSN) * 2)     // hi+lo bf16 for x and W_s (~101 MB)

union BfPack { __bf16 h[4]; unsigned long long u; };

// async global->LDS builtin wants v4i pointers in explicit address spaces
typedef int v4i __attribute__((vector_size(16)));
typedef __attribute__((address_space(1))) v4i av4i_g;   // global
typedef __attribute__((address_space(3))) v4i av4i_l;   // LDS

__device__ __forceinline__ void async_cp16(const __bf16* g, __bf16* l) {
    // 16B global -> LDS, tracked by ASYNCcnt
    __builtin_amdgcn_global_load_async_to_lds_b128((av4i_g*)g, (av4i_l*)l, 0, 0);
}

__device__ __forceinline__ v16bf lds_frag16(const __bf16* p0, const __bf16* p1) {
    v8bf a = *(const v8bf*)p0;
    v8bf b = *(const v8bf*)p1;
    v16bf r;
#pragma unroll
    for (int i = 0; i < 8; ++i) { r[i] = a[i]; r[i + 8] = b[i]; }
    return r;
}

__device__ __forceinline__ void split4(const float4 v, BfPack& ph, BfPack& pl) {
    ph.h[0] = (__bf16)v.x; pl.h[0] = (__bf16)(v.x - (float)ph.h[0]);
    ph.h[1] = (__bf16)v.y; pl.h[1] = (__bf16)(v.y - (float)ph.h[1]);
    ph.h[2] = (__bf16)v.z; pl.h[2] = (__bf16)(v.z - (float)ph.h[2]);
    ph.h[3] = (__bf16)v.w; pl.h[3] = (__bf16)(v.w - (float)ph.h[3]);
}

// ---------------------------------------------------------------------------
// Pre-pass: split f32 x and W_s into bf16 hi/lo arrays in workspace.
// ws layout (bf16 elems): xhi[XN] | xlo[XN] | wshi[WSN] | wslo[WSN]
// ---------------------------------------------------------------------------
__global__ __launch_bounds__(256)
void split_bf16_kernel(const float* __restrict__ x, const float* __restrict__ Ws,
                       __bf16* __restrict__ ws) {
    const size_t f  = (size_t)blockIdx.x * 256 + threadIdx.x;  // float4 index
    const size_t XF = XN / 4;
    const float* src;
    __bf16 *dhi, *dlo;
    if (f < XF) {
        const size_t e = f * 4;
        src = x + e;  dhi = ws + e;           dlo = ws + XN + e;
    } else {
        const size_t e = (f - XF) * 4;
        src = Ws + e; dhi = ws + 2 * XN + e;  dlo = ws + 2 * XN + WSN + e;
    }
    const float4 v = *(const float4*)src;
    BfPack ph, pl;
    split4(v, ph, pl);
    *(unsigned long long*)dhi = ph.u;
    *(unsigned long long*)dlo = pl.u;
}

// ---------------------------------------------------------------------------
// Main GEMM (pre-split inputs): async global->LDS staging, zero conversion
// VALU in the hot loop.  out[b,n] = sum_c Wd[n,c] * (sum_hw x*Ws) + Wb[n]
// ---------------------------------------------------------------------------
__global__ __launch_bounds__(256)
void conv_encoder_wmma_pre(const __bf16* __restrict__ ws, const float* __restrict__ Wd,
                           const float* __restrict__ Wb, float* __restrict__ out) {
    __shared__ __align__(16) __bf16 Ahi[CHANNELS * LDS_STRIDE];
    __shared__ __align__(16) __bf16 Alo[CHANNELS * LDS_STRIDE];
    __shared__ __align__(16) __bf16 Bhi[NTILE   * LDS_STRIDE];
    __shared__ __align__(16) __bf16 Blo[NTILE   * LDS_STRIDE];
    __shared__ float OutAcc[NTILE];

    const int tid = threadIdx.x;
    const int b   = blockIdx.y;
    const int n0  = blockIdx.x * NTILE;

    if (tid < NTILE) OutAcc[tid] = 0.0f;

    const int lane = tid & 31;
    const int wave = tid >> 5;
    const int wm   = wave & 3;       // c-block (4 x 16 rows)
    const int wn   = wave >> 2;      // n-block (2 x 64 cols)
    const int lsel = lane >> 4;
    const int lrow = lane & 15;

    const __bf16* __restrict__ xAhi = ws +                 (size_t)(b * CHANNELS) * HW;
    const __bf16* __restrict__ xAlo = ws + XN +            (size_t)(b * CHANNELS) * HW;
    const __bf16* __restrict__ wBhi = ws + 2 * XN +        (size_t)n0 * HW;
    const __bf16* __restrict__ wBlo = ws + 2 * XN + WSN +  (size_t)n0 * HW;

    v8f acc[4] = {v8f{}, v8f{}, v8f{}, v8f{}};

    for (int k0 = 0; k0 < HW; k0 += KTILE) {
        // ---- async stage A (64x64) : 512 x 16B chunks per array ----
#pragma unroll
        for (int i = 0; i < 2; ++i) {
            const int idx = tid + i * 256;        // 0..511
            const int row = idx >> 3;             // 0..63
            const int ch  = (idx & 7) << 3;       // elem offset 0..56
            const size_t g = (size_t)row * HW + k0 + ch;
            const int    o = row * LDS_STRIDE + ch;
            async_cp16(xAhi + g, &Ahi[o]);
            async_cp16(xAlo + g, &Alo[o]);
        }
        // ---- async stage B (128x64) : 1024 x 16B chunks per array ----
#pragma unroll
        for (int i = 0; i < 4; ++i) {
            const int idx = tid + i * 256;        // 0..1023
            const int row = idx >> 3;             // 0..127
            const int ch  = (idx & 7) << 3;
            const size_t g = (size_t)row * HW + k0 + ch;
            const int    o = row * LDS_STRIDE + ch;
            async_cp16(wBhi + g, &Bhi[o]);
            async_cp16(wBlo + g, &Blo[o]);
        }
        asm volatile("s_wait_asynccnt 0" ::: "memory");
        __syncthreads();

        // ---- compute: 2 K-steps x 4 N-blocks x 3 split products ----
#pragma unroll
        for (int ks = 0; ks < KTILE; ks += 32) {
            const int ab = (wm * 16 + lrow) * LDS_STRIDE + ks + (lsel << 4);
            const v16bf ahi = lds_frag16(&Ahi[ab], &Ahi[ab + 8]);
            const v16bf alo = lds_frag16(&Alo[ab], &Alo[ab + 8]);
#pragma unroll
            for (int j = 0; j < 4; ++j) {
                const int bb = (wn * 64 + j * 16 + lrow) * LDS_STRIDE + ks + (lsel << 4);
                const v16bf bh = lds_frag16(&Bhi[bb], &Bhi[bb + 8]);
                const v16bf bl = lds_frag16(&Blo[bb], &Blo[bb + 8]);
                acc[j] = __builtin_amdgcn_wmma_f32_16x16x32_bf16(
                    false, ahi, false, bh, (short)0, acc[j], false, false);
                acc[j] = __builtin_amdgcn_wmma_f32_16x16x32_bf16(
                    false, ahi, false, bl, (short)0, acc[j], false, false);
                acc[j] = __builtin_amdgcn_wmma_f32_16x16x32_bf16(
                    false, alo, false, bh, (short)0, acc[j], false, false);
            }
        }
        __syncthreads();
    }

    // ---- epilogue: weight by Wd[n,c], reduce over c, add bias ----
    const int c_base = wm * 16 + (lsel << 3);
#pragma unroll
    for (int j = 0; j < 4; ++j) {
        const int nloc = wn * 64 + j * 16 + lrow;
        const float* __restrict__ wd = Wd + (size_t)(n0 + nloc) * CHANNELS + c_base;
        float s = 0.0f;
#pragma unroll
        for (int r = 0; r < 8; ++r) s += acc[j][r] * wd[r];
        atomicAdd(&OutAcc[nloc], s);   // ds_add_f32
    }
    __syncthreads();

    if (tid < NTILE) {
        const int n = n0 + tid;
        out[(size_t)b * NEURONS + n] = OutAcc[tid] + Wb[n];
    }
}

// ---------------------------------------------------------------------------
// Fallback (workspace too small): fused in-loop conversion (round 1 kernel).
// ---------------------------------------------------------------------------
__global__ __launch_bounds__(256)
void conv_encoder_fused_wmma(const float* __restrict__ x,
                             const float* __restrict__ Ws,
                             const float* __restrict__ Wd,
                             const float* __restrict__ Wb,
                             float* __restrict__ out) {
    __shared__ __align__(16) __bf16 Ahi[CHANNELS * LDS_STRIDE];
    __shared__ __align__(16) __bf16 Alo[CHANNELS * LDS_STRIDE];
    __shared__ __align__(16) __bf16 Bhi[NTILE   * LDS_STRIDE];
    __shared__ __align__(16) __bf16 Blo[NTILE   * LDS_STRIDE];
    __shared__ float OutAcc[NTILE];

    const int tid = threadIdx.x;
    const int b   = blockIdx.y;
    const int n0  = blockIdx.x * NTILE;

    if (tid < NTILE) OutAcc[tid] = 0.0f;

    const int lane = tid & 31;
    const int wave = tid >> 5;
    const int wm   = wave & 3;
    const int wn   = wave >> 2;
    const int lsel = lane >> 4;
    const int lrow = lane & 15;

    const float* __restrict__ xA = x  + (size_t)(b * CHANNELS) * HW;
    const float* __restrict__ wB = Ws + (size_t)n0 * HW;

    v8f acc[4] = {v8f{}, v8f{}, v8f{}, v8f{}};

    for (int k0 = 0; k0 < HW; k0 += KTILE) {
#pragma unroll
        for (int i = 0; i < 4; ++i) {
            const int idx = tid + i * 256;
            const int row = idx >> 4;
            const int kc  = (idx & 15) << 2;
            const float4 v = *(const float4*)(xA + (size_t)row * HW + (k0 + kc));
            BfPack ph, pl;
            split4(v, ph, pl);
            const int o = row * LDS_STRIDE + kc;
            *(unsigned long long*)&Ahi[o] = ph.u;
            *(unsigned long long*)&Alo[o] = pl.u;
        }
#pragma unroll
        for (int i = 0; i < 8; ++i) {
            const int idx = tid + i * 256;
            const int row = idx >> 4;
            const int kc  = (idx & 15) << 2;
            const float4 v = *(const float4*)(wB + (size_t)row * HW + (k0 + kc));
            BfPack ph, pl;
            split4(v, ph, pl);
            const int o = row * LDS_STRIDE + kc;
            *(unsigned long long*)&Bhi[o] = ph.u;
            *(unsigned long long*)&Blo[o] = pl.u;
        }
        __syncthreads();

#pragma unroll
        for (int ks = 0; ks < KTILE; ks += 32) {
            const int ab = (wm * 16 + lrow) * LDS_STRIDE + ks + (lsel << 4);
            const v16bf ahi = lds_frag16(&Ahi[ab], &Ahi[ab + 8]);
            const v16bf alo = lds_frag16(&Alo[ab], &Alo[ab + 8]);
#pragma unroll
            for (int j = 0; j < 4; ++j) {
                const int bb = (wn * 64 + j * 16 + lrow) * LDS_STRIDE + ks + (lsel << 4);
                const v16bf bh = lds_frag16(&Bhi[bb], &Bhi[bb + 8]);
                const v16bf bl = lds_frag16(&Blo[bb], &Blo[bb + 8]);
                acc[j] = __builtin_amdgcn_wmma_f32_16x16x32_bf16(
                    false, ahi, false, bh, (short)0, acc[j], false, false);
                acc[j] = __builtin_amdgcn_wmma_f32_16x16x32_bf16(
                    false, ahi, false, bl, (short)0, acc[j], false, false);
                acc[j] = __builtin_amdgcn_wmma_f32_16x16x32_bf16(
                    false, alo, false, bh, (short)0, acc[j], false, false);
            }
        }
        __syncthreads();
    }

    const int c_base = wm * 16 + (lsel << 3);
#pragma unroll
    for (int j = 0; j < 4; ++j) {
        const int nloc = wn * 64 + j * 16 + lrow;
        const float* __restrict__ wd = Wd + (size_t)(n0 + nloc) * CHANNELS + c_base;
        float s = 0.0f;
#pragma unroll
        for (int r = 0; r < 8; ++r) s += acc[j][r] * wd[r];
        atomicAdd(&OutAcc[nloc], s);
    }
    __syncthreads();

    if (tid < NTILE) {
        const int n = n0 + tid;
        out[(size_t)b * NEURONS + n] = OutAcc[tid] + Wb[n];
    }
}

extern "C" void kernel_launch(void* const* d_in, const int* in_sizes, int n_in,
                              void* d_out, int out_size, void* d_ws, size_t ws_size,
                              hipStream_t stream) {
    (void)in_sizes; (void)n_in; (void)out_size;
    const float* x  = (const float*)d_in[0];   // (32, 64, 64, 64)
    const float* Ws = (const float*)d_in[1];   // (4096, 64, 64)
    const float* Wd = (const float*)d_in[2];   // (4096, 64, 1, 1)
    const float* Wb = (const float*)d_in[3];   // (1, 4096)
    float* out = (float*)d_out;                // (32, 4096)

    dim3 grid(NEURONS / NTILE, BATCH);         // 32 x 32 workgroups

    if (ws_size >= WS_BYTES) {
        const int nblk = (int)((XN / 4 + WSN / 4) / 256);   // 24576
        split_bf16_kernel<<<nblk, 256, 0, stream>>>(x, Ws, (__bf16*)d_ws);
        conv_encoder_wmma_pre<<<grid, 256, 0, stream>>>(
            (const __bf16*)d_ws, Wd, Wb, out);
    } else {
        conv_encoder_fused_wmma<<<grid, 256, 0, stream>>>(x, Ws, Wd, Wb, out);
    }
}
